// GAT_layer_67619965108552
// MI455X (gfx1250) — matrix operand order repaired and checked by reading, compiled, not run on hardware
//
#include <hip/hip_runtime.h>

#define N_NODES 8192
#define F_IN    512
#define F_OUT   256
#define HALF_N  4096

typedef __attribute__((ext_vector_type(16))) _Float16 v16h;
typedef __attribute__((ext_vector_type(8)))  float    v8f;

union Frag {
  v16h v;
  _Float16 h[16];
  uint4 q[2];
};

__device__ __forceinline__ float lrelu(float v)    { return fmaxf(v, 0.01f * v); }
__device__ __forceinline__ float sigmoidf(float v) { return 1.0f / (1.0f + __expf(-v)); }

// ---------------------------------------------------------------------------
// K0: weight (512x256 f32, row-major) -> Wt (256x512 f16, K-contiguous rows)
// ---------------------------------------------------------------------------
__global__ __launch_bounds__(256) void wt_convert(const float* __restrict__ W,
                                                  _Float16* __restrict__ Wt) {
  int t = blockIdx.x * 256 + threadIdx.x;   // coalesced read of W[k][n]
  int k = t >> 8;
  int n = t & 255;
  Wt[(size_t)n * F_IN + k] = (_Float16)W[t];
}

// ---------------------------------------------------------------------------
// K1: H = x @ weight  (8192x512 @ 512x256), f16 WMMA, f32 accumulate.
// Block = 16 output rows, 8 waves x 2 N-tiles (16 cols each) = 256 cols.
// ---------------------------------------------------------------------------
__global__ __launch_bounds__(256) void gemm_xw(const float* __restrict__ x,
                                               const _Float16* __restrict__ Wt,
                                               float* __restrict__ H) {
  const int i0   = blockIdx.x * 16;
  const int tid  = threadIdx.x;
  const int lane = tid & 31;
  const int w    = tid >> 5;
  const int n0   = w * 2;
  const int ac   = (lane >> 4) * 8;          // A K-chunk base (halves)

  v8f acc0 = {}; v8f acc1 = {};
  const float* __restrict__ xrow = x + (size_t)(i0 + (lane & 15)) * F_IN;
  const uint4* __restrict__ b0p =
      (const uint4*)(Wt + (size_t)(n0 * 16 + (lane & 15)) * F_IN + (lane >> 4) * 8);
  const uint4* __restrict__ b1p =
      (const uint4*)(Wt + (size_t)((n0 + 1) * 16 + (lane & 15)) * F_IN + (lane >> 4) * 8);

  for (int kt = 0; kt < F_IN / 32; ++kt) {
    const int kb = kt * 32;
    Frag A, B0, B1;
#pragma unroll
    for (int q2 = 0; q2 < 8; ++q2) A.h[q2]     = (_Float16)xrow[kb + ac + q2];
#pragma unroll
    for (int q2 = 0; q2 < 8; ++q2) A.h[8 + q2] = (_Float16)xrow[kb + 16 + ac + q2];
    B0.q[0] = b0p[kt * 4]; B0.q[1] = b0p[kt * 4 + 2];
    B1.q[0] = b1p[kt * 4]; B1.q[1] = b1p[kt * 4 + 2];
    acc0 = __builtin_amdgcn_wmma_f32_16x16x32_f16(false, A.v, false, B0.v, (short)0, acc0, false, false);
    acc1 = __builtin_amdgcn_wmma_f32_16x16x32_f16(false, A.v, false, B1.v, (short)0, acc1, false, false);
  }
  const int mh = (lane >> 4) * 8;
  const int nc = lane & 15;
#pragma unroll
  for (int rr = 0; rr < 8; ++rr) {
    const int M = rr + mh;
    H[(size_t)(i0 + M) * F_OUT + n0 * 16 + nc]       = acc0[rr];
    H[(size_t)(i0 + M) * F_OUT + (n0 + 1) * 16 + nc] = acc1[rr];
  }
}

// ---------------------------------------------------------------------------
// K2: transpose H (8192x256 f32) -> HbT (256x8192 f16) via LDS tile.
// ---------------------------------------------------------------------------
__global__ __launch_bounds__(256) void transpose_h(const float* __restrict__ H,
                                                   _Float16* __restrict__ HbT) {
  __shared__ _Float16 tile[32][33];
  const int bx = blockIdx.x;   // H row tile
  const int by = blockIdx.y;   // H col tile
  const int tx = threadIdx.x;  // 0..31
  const int ty = threadIdx.y;  // 0..7
#pragma unroll
  for (int y = ty; y < 32; y += 8)
    tile[y][tx] = (_Float16)H[(size_t)(bx * 32 + y) * F_OUT + by * 32 + tx];
  __syncthreads();
#pragma unroll
  for (int y = ty; y < 32; y += 8)
    HbT[(size_t)(by * 32 + y) * N_NODES + bx * 32 + tx] = tile[tx][y];
}

// ---------------------------------------------------------------------------
// K3: s1[i] = H[i,:].a1 ; s2[i] = H[i,:].a2   (one wave32 per row)
// ---------------------------------------------------------------------------
__global__ __launch_bounds__(256) void svec(const float* __restrict__ H,
                                            const float* __restrict__ att,
                                            float* __restrict__ s1,
                                            float* __restrict__ s2) {
  const int tid  = threadIdx.x;
  const int lane = tid & 31;
  const int row  = blockIdx.x * 8 + (tid >> 5);
  const float* __restrict__ h  = H + (size_t)row * F_OUT + lane * 8;
  const float* __restrict__ a1 = att + lane * 8;
  const float* __restrict__ a2 = att + F_OUT + lane * 8;
  float d1 = 0.0f, d2 = 0.0f;
#pragma unroll
  for (int q = 0; q < 8; ++q) {
    const float hv = h[q];
    d1 += hv * a1[q];
    d2 += hv * a2[q];
  }
#pragma unroll
  for (int off = 16; off > 0; off >>= 1) {
    d1 += __shfl_xor(d1, off, 32);
    d2 += __shfl_xor(d2, off, 32);
  }
  if (lane == 0) { s1[row] = d1; s2[row] = d2; }
}

// ---------------------------------------------------------------------------
// K4: build softmax weight tables.
//   u[k] = exp(lrelu(s1[k]+s2[k]) - max over the (2i,2i+1) pair)   [top rows]
//   e[j] = exp(lrelu(s1[(2j)%N]+s2[(2j+1)%N]) - global max)        [bottom rows]
// Normalization cancels the shift, so this matches the reference softmax.
// ---------------------------------------------------------------------------
__global__ __launch_bounds__(1024) void prep(const float* __restrict__ s1,
                                             const float* __restrict__ s2,
                                             float* __restrict__ u,
                                             float* __restrict__ e) {
  __shared__ float red[1024];
  const int tid = threadIdx.x;
  float lt[8];
  float m = -3.0e38f;
#pragma unroll
  for (int q = 0; q < 8; ++q) {
    const int j = tid + q * 1024;
    const float a = lrelu(s1[j] + s2[j]);
    const int   p = j ^ 1;
    const float b = lrelu(s1[p] + s2[p]);
    u[j] = __expf(a - fmaxf(a, b));
    const int j1 = (2 * j)     & (N_NODES - 1);
    const int j2 = (2 * j + 1) & (N_NODES - 1);
    const float v = lrelu(s1[j1] + s2[j2]);
    lt[q] = v;
    m = fmaxf(m, v);
  }
  red[tid] = m;
  __syncthreads();
  for (int s = 512; s > 0; s >>= 1) {
    if (tid < s) red[tid] = fmaxf(red[tid], red[tid + s]);
    __syncthreads();
  }
  const float M = red[0];
#pragma unroll
  for (int q = 0; q < 8; ++q) e[tid + q * 1024] = __expf(lt[q] - M);
}

// ---------------------------------------------------------------------------
// K5: fused attention: out = sigmoid( (P @ H) / Z ), P built on the fly from
// adj + weight tables; single pass over adj (the bandwidth-critical stream).
// Block = 16 rows; 8 waves x 2 N-tiles; 256 K-steps of 32 columns.
// P tile double-buffered in LDS (1 barrier/step), reloaded in A-matrix layout.
// ---------------------------------------------------------------------------
__global__ __launch_bounds__(256) void attn_pv(const int* __restrict__ adj,
                                               const _Float16* __restrict__ HbT,
                                               const float* __restrict__ u,
                                               const float* __restrict__ e,
                                               float* __restrict__ out) {
  __shared__ unsigned int Pw[2][16][16];   // 16x32 f16 tile as packed pairs
  __shared__ float zred[256];
  __shared__ float Zrow[16];
  __shared__ float urow[2][16];

  const int  i0   = blockIdx.x * 16;
  const bool top  = (i0 < HALF_N);
  const int  tid  = threadIdx.x;
  const int  r    = tid >> 4;              // row within tile this thread fills
  const int  jl   = (tid & 15) << 1;       // col pair within 32-wide tile
  const int  lane = tid & 31;
  const int  w    = tid >> 5;
  const int  n0   = w * 2;

  if (top && tid < 16) {
    urow[0][tid] = u[2 * (i0 + tid)];
    urow[1][tid] = u[2 * (i0 + tid) + 1];
  }
  __syncthreads();
  const float uw0 = urow[0][r];
  const float uw1 = urow[1][r];

  v8f acc0 = {}; v8f acc1 = {};
  float zpart = 0.0f;

  const int ar  = lane & 15;               // A row for this lane
  const int aci = lane >> 4;               // A K-chunk select
  const uint4* __restrict__ b0p =
      (const uint4*)(HbT + (size_t)(n0 * 16 + (lane & 15)) * N_NODES + (lane >> 4) * 8);
  const uint4* __restrict__ b1p =
      (const uint4*)(HbT + (size_t)((n0 + 1) * 16 + (lane & 15)) * N_NODES + (lane >> 4) * 8);
  const int* __restrict__ adjrow = adj + (size_t)(i0 + r) * N_NODES + jl;

  for (int jt = 0; jt < N_NODES / 32; ++jt) {
    const int buf = jt & 1;
    const int j   = jt * 32 + jl;

    const int2 a2 = *(const int2*)(adjrow + jt * 32);
    float w0, w1;
    if (top) {
      const float uv = (j < HALF_N) ? uw0 : uw1;   // j, j+1 on same side
      w0 = (a2.x > 0) ? uv : 0.0f;
      w1 = (a2.y > 0) ? uv : 0.0f;
    } else {
      const float2 ev = *(const float2*)(e + j);
      w0 = (a2.x > 0) ? ev.x : 0.0f;
      w1 = (a2.y > 0) ? ev.y : 0.0f;
    }
    zpart += w0 + w1;
    union { _Float16 h[2]; unsigned int p; } pk;
    pk.h[0] = (_Float16)w0; pk.h[1] = (_Float16)w1;
    Pw[buf][r][jl >> 1] = pk.p;
    __syncthreads();                          // P tile visible to all waves

    Frag A, B0, B1;
    const uint4* arow = (const uint4*)&Pw[buf][ar][0];
    A.q[0] = arow[aci];                       // K 0..7  / 8..15
    A.q[1] = arow[2 + aci];                   // K 16..23 / 24..31
    B0.q[0] = b0p[jt * 4]; B0.q[1] = b0p[jt * 4 + 2];
    B1.q[0] = b1p[jt * 4]; B1.q[1] = b1p[jt * 4 + 2];

    acc0 = __builtin_amdgcn_wmma_f32_16x16x32_f16(false, A.v, false, B0.v, (short)0, acc0, false, false);
    acc1 = __builtin_amdgcn_wmma_f32_16x16x32_f16(false, A.v, false, B1.v, (short)0, acc1, false, false);
  }

  // per-row normalizer Z
  zred[tid] = zpart;
  __syncthreads();
  if (tid < 16) {
    float z = 0.0f;
#pragma unroll
    for (int t2 = 0; t2 < 16; ++t2) z += zred[tid * 16 + t2];
    Zrow[tid] = z;
  }
  __syncthreads();

  const int mh = (lane >> 4) * 8;
  const int nc = lane & 15;
#pragma unroll
  for (int rr = 0; rr < 8; ++rr) {
    const int M = rr + mh;
    const float zv  = Zrow[M];
    const float inv = (zv > 0.0f) ? (1.0f / zv) : 0.0f;
    out[(size_t)(i0 + M) * F_OUT + n0 * 16 + nc]       = sigmoidf(acc0[rr] * inv);
    out[(size_t)(i0 + M) * F_OUT + (n0 + 1) * 16 + nc] = sigmoidf(acc1[rr] * inv);
  }
}

// ---------------------------------------------------------------------------
extern "C" void kernel_launch(void* const* d_in, const int* in_sizes, int n_in,
                              void* d_out, int out_size, void* d_ws, size_t ws_size,
                              hipStream_t stream) {
  (void)in_sizes; (void)n_in; (void)out_size; (void)ws_size;
  const float* x      = (const float*)d_in[0];
  const float* weight = (const float*)d_in[1];
  const float* attv   = (const float*)d_in[2];
  const int*   adj    = (const int*)d_in[3];
  float* out = (float*)d_out;

  char* ws = (char*)d_ws;
  float*    H   = (float*)ws;     ws += (size_t)N_NODES * F_OUT * sizeof(float);
  _Float16* HbT = (_Float16*)ws;  ws += (size_t)F_OUT * N_NODES * sizeof(_Float16);
  _Float16* Wt  = (_Float16*)ws;  ws += (size_t)F_OUT * F_IN * sizeof(_Float16);
  float* s1 = (float*)ws;         ws += (size_t)N_NODES * sizeof(float);
  float* s2 = (float*)ws;         ws += (size_t)N_NODES * sizeof(float);
  float* u  = (float*)ws;         ws += (size_t)N_NODES * sizeof(float);
  float* e  = (float*)ws;         ws += (size_t)N_NODES * sizeof(float);

  wt_convert<<<(F_IN * F_OUT) / 256, 256, 0, stream>>>(weight, Wt);
  gemm_xw<<<N_NODES / 16, 256, 0, stream>>>(x, Wt, H);
  transpose_h<<<dim3(N_NODES / 32, F_OUT / 32), dim3(32, 8), 0, stream>>>(H, HbT);
  svec<<<N_NODES / 8, 256, 0, stream>>>(H, attv, s1, s2);
  prep<<<1, 1024, 0, stream>>>(s1, s2, u, e);
  attn_pv<<<N_NODES / 16, 256, 0, stream>>>(adj, HbT, u, e, out);
}
